// LayerNormLSTM_13529146983013
// MI455X (gfx1250) — matrix operand order repaired and checked by reading, not compiled, not run on hardware
//
#include <hip/hip_runtime.h>
#include <hip/hip_bf16.h>
#include <math.h>

// ---------------------------------------------------------------------------
// LayerNorm-LSTM for MI455X (gfx1250): bf16 WMMA GEMMs + fused LN/gate cell.
// B=64, T=1024, D=H=512, 4H=2048, L=2. All GEMMs have K=512, N=2048.
// Weights pre-converted to bf16; recurrent path loads pure bf16; GEMM K-loop
// is double-buffered so L2 latency overlaps with WMMA issue.
// ---------------------------------------------------------------------------

typedef __bf16 v16bf __attribute__((ext_vector_type(16)));
typedef __bf16 v8bf  __attribute__((ext_vector_type(8)));
typedef float  v8f   __attribute__((ext_vector_type(8)));

#define GN 2048   // 4*H (gate width)
#define GK 512    // D == H
#define KITERS (GK / 32)
#define LN_EPS 1e-5f

// ---------------------------------------------------------------------------
// fp32 -> bf16 elementwise conversion (n must be a multiple of 8)
// ---------------------------------------------------------------------------
__global__ __launch_bounds__(256) void f32_to_bf16_kernel(
    const float* __restrict__ in, __bf16* __restrict__ out, int n)
{
  const int i = (blockIdx.x * blockDim.x + threadIdx.x) * 8;
  if (i >= n) return;
  const float4 f0 = *(const float4*)(in + i);
  const float4 f1 = *(const float4*)(in + i + 4);
  v8bf o;
  o[0]=(__bf16)f0.x; o[1]=(__bf16)f0.y; o[2]=(__bf16)f0.z; o[3]=(__bf16)f0.w;
  o[4]=(__bf16)f1.x; o[5]=(__bf16)f1.y; o[6]=(__bf16)f1.z; o[7]=(__bf16)f1.w;
  *(v8bf*)(out + i) = o;
}

// ---------------------------------------------------------------------------
// Fragment loaders.
// A fragment 16x32 (per ISA): lanes 0-15 K={ka..ka+7, ka+16..ka+23} with
// ka = k0 + 8*half; elements interleaved 8/8.
// ---------------------------------------------------------------------------
template<bool A_BF16>
__device__ inline v16bf load_a_frag(const void* Av, long aoff, int ka) {
  if constexpr (A_BF16) {
    const __bf16* Arow = (const __bf16*)Av + aoff;
    const v8bf lo = *(const v8bf*)(Arow + ka);
    const v8bf hi = *(const v8bf*)(Arow + ka + 16);
    return __builtin_shufflevector(lo, hi, 0,1,2,3,4,5,6,7,8,9,10,11,12,13,14,15);
  } else {
    const float* Arow = (const float*)Av + aoff;
    const float4 f0 = *(const float4*)(Arow + ka);
    const float4 f1 = *(const float4*)(Arow + ka + 4);
    const float4 f2 = *(const float4*)(Arow + ka + 16);
    const float4 f3 = *(const float4*)(Arow + ka + 20);
    v16bf a;
    a[0]=(__bf16)f0.x;  a[1]=(__bf16)f0.y;  a[2]=(__bf16)f0.z;  a[3]=(__bf16)f0.w;
    a[4]=(__bf16)f1.x;  a[5]=(__bf16)f1.y;  a[6]=(__bf16)f1.z;  a[7]=(__bf16)f1.w;
    a[8]=(__bf16)f2.x;  a[9]=(__bf16)f2.y;  a[10]=(__bf16)f2.z; a[11]=(__bf16)f2.w;
    a[12]=(__bf16)f3.x; a[13]=(__bf16)f3.y; a[14]=(__bf16)f3.z; a[15]=(__bf16)f3.w;
    return a;
  }
}

// ---------------------------------------------------------------------------
// WMMA GEMM:  C[m,n] = sum_k A_row(m)[k] * W[n*GK + k]   (i.e. A @ W^T)
// W is bf16 (preconverted). A is fp32 (bulk) or bf16 (recurrent) via template.
// A row address: A + (m % Bdim)*strideB + (t0 + m / Bdim)*strideT
// One wave computes a 16(M) x 64(N) strip: A-frag reused across 4 WMMAs.
// K-loop is software-pipelined (double-buffered) to hide L2 latency.
// blockIdx.y selects between two independent GEMM problems (layer-1 pair).
// ---------------------------------------------------------------------------
template<bool A_BF16>
__global__ __launch_bounds__(256) void wmma_gemm_kernel(
    const void* __restrict__ A0v, const __bf16* __restrict__ W0, float* __restrict__ C0,
    const void* __restrict__ A1v, const __bf16* __restrict__ W1, float* __restrict__ C1,
    int M, int Bdim, int strideB, int strideT, int t0)
{
  const void* Av = A0v; const __bf16* W = W0; float* C = C0;
  if (blockIdx.y == 1) { Av = A1v; W = W1; C = C1; }

  const int wave    = (blockIdx.x * blockDim.x + threadIdx.x) >> 5;
  const int nTilesN = GN / 64;           // 32
  const int nTilesM = M >> 4;
  if (wave >= nTilesM * nTilesN) return; // wave-uniform: EXEC stays all-ones for WMMA

  const int tileM = wave / nTilesN;
  const int tileN = wave % nTilesN;
  const int lane  = threadIdx.x & 31;
  const int half  = lane >> 4;
  const int r     = lane & 15;

  // A row for this lane (A layout: row = lane%16 in each half)
  const int  mrow = (tileM << 4) + r;
  const int  bb   = mrow % Bdim;
  const int  tt   = t0 + mrow / Bdim;
  const long aoff = (long)bb * strideB + (long)tt * strideT;

  // W rows (B operand columns) for the 4 N-subtiles
  const int ncol = (tileN << 6) + r;
  const __bf16* Wr0 = W + (long)(ncol +  0) * GK;
  const __bf16* Wr1 = W + (long)(ncol + 16) * GK;
  const __bf16* Wr2 = W + (long)(ncol + 32) * GK;
  const __bf16* Wr3 = W + (long)(ncol + 48) * GK;

  v8f acc0 = {}, acc1 = {}, acc2 = {}, acc3 = {};

  // Double-buffered fragments (fully unrolled loop -> stays in registers)
  v16bf abuf[2];
  v16bf bbuf[2][4];
  {
    const int ka = (half << 3);
    const int kb = (half << 3);
    abuf[0]    = load_a_frag<A_BF16>(Av, aoff, ka);
    bbuf[0][0] = load_a_frag<true>(Wr0, 0, kb);
    bbuf[0][1] = load_a_frag<true>(Wr1, 0, kb);
    bbuf[0][2] = load_a_frag<true>(Wr2, 0, kb);
    bbuf[0][3] = load_a_frag<true>(Wr3, 0, kb);
  }

#pragma unroll
  for (int it = 0; it < KITERS; ++it) {
    const int cur = it & 1;
    const int nxt = cur ^ 1;
    if (it + 1 < KITERS) {
      // Issue ALL of next iteration's loads before consuming this one:
      // the s_wait before the WMMAs then overlaps with these in-flight loads.
      const int kn = (it + 1) << 5;
      const int ka = kn + (half << 3);
      const int kb = kn + (half << 3);
      abuf[nxt]    = load_a_frag<A_BF16>(Av, aoff, ka);
      bbuf[nxt][0] = load_a_frag<true>(Wr0, 0, kb);
      bbuf[nxt][1] = load_a_frag<true>(Wr1, 0, kb);
      bbuf[nxt][2] = load_a_frag<true>(Wr2, 0, kb);
      bbuf[nxt][3] = load_a_frag<true>(Wr3, 0, kb);
    }
    acc0 = __builtin_amdgcn_wmma_f32_16x16x32_bf16(false, abuf[cur], false, bbuf[cur][0], (short)0, acc0, false, false);
    acc1 = __builtin_amdgcn_wmma_f32_16x16x32_bf16(false, abuf[cur], false, bbuf[cur][1], (short)0, acc1, false, false);
    acc2 = __builtin_amdgcn_wmma_f32_16x16x32_bf16(false, abuf[cur], false, bbuf[cur][2], (short)0, acc2, false, false);
    acc3 = __builtin_amdgcn_wmma_f32_16x16x32_bf16(false, abuf[cur], false, bbuf[cur][3], (short)0, acc3, false, false);
  }

  // C/D layout: row = v + 8*half, col = lane%16 (per 16-wide subtile)
  const long crow0 = (long)((tileM << 4) + (half << 3)) * GN;
  const int  cb    = (tileN << 6) + r;
#pragma unroll
  for (int v = 0; v < 8; ++v) {
    const long ro = crow0 + (long)v * GN;
    C[ro + cb +  0] = acc0[v];
    C[ro + cb + 16] = acc1[v];
    C[ro + cb + 32] = acc2[v];
    C[ro + cb + 48] = acc3[v];
  }
}

// ---------------------------------------------------------------------------
// Block-wide sum: wave32 shuffle butterfly + one 8-partial LDS broadcast.
// (2 barriers per call instead of an 8-level LDS tree.)
// ---------------------------------------------------------------------------
__device__ inline float block_sum(float v) {
#pragma unroll
  for (int off = 16; off > 0; off >>= 1)
    v += __shfl_xor(v, off, 32);
  __shared__ float red[8];
  const int wid  = threadIdx.x >> 5;
  const int lane = threadIdx.x & 31;
  if (lane == 0) red[wid] = v;
  __syncthreads();
  float s = red[0] + red[1] + red[2] + red[3]
          + red[4] + red[5] + red[6] + red[7];
  __syncthreads();
  return s;
}

__device__ inline float sigmoidf(float x) { return 1.0f / (1.0f + __expf(-x)); }

// ---------------------------------------------------------------------------
// Fused cell kernel: one block per batch row b (grid=64, block=256).
//   gates = LN(ig_raw; lni) + LN(hg_raw; lnh)        (width 2048)
//   i,f,g,o = split(gates); c' = LN(f*c + i*g; lnc)  (width 512)
//   h' = o * tanh(c');  writes c (f32), h (f32 + bf16) and optional strided
//   extra f32 h destination (final output tensor for layer 1).
// ---------------------------------------------------------------------------
__global__ __launch_bounds__(256) void cell_kernel(
    const float* __restrict__ igr, const float* __restrict__ hgr,
    const float* __restrict__ lni_g, const float* __restrict__ lni_b,
    const float* __restrict__ lnh_g, const float* __restrict__ lnh_b,
    const float* __restrict__ lnc_g, const float* __restrict__ lnc_b,
    float* __restrict__ cbuf, float* __restrict__ hbuf, __bf16* __restrict__ hbf,
    float* __restrict__ hx, long hx_stride)
{
  const int tid = threadIdx.x;
  const int b   = blockIdx.x;
  const float* ig = igr + (long)b * GN;
  const float* hg = hgr + (long)b * GN;

  float iv[8], hv[8];
  float si = 0.f, sh = 0.f;
#pragma unroll
  for (int j = 0; j < 8; ++j) {
    const int idx = tid + 256 * j;
    iv[j] = ig[idx]; hv[j] = hg[idx];
    si += iv[j]; sh += hv[j];
  }
  const float mi = block_sum(si) * (1.0f / GN);
  const float mh = block_sum(sh) * (1.0f / GN);
  float qi = 0.f, qh = 0.f;
#pragma unroll
  for (int j = 0; j < 8; ++j) {
    float d = iv[j] - mi; qi += d * d;
    d = hv[j] - mh;       qh += d * d;
  }
  const float ri = rsqrtf(block_sum(qi) * (1.0f / GN) + LN_EPS);
  const float rh = rsqrtf(block_sum(qh) * (1.0f / GN) + LN_EPS);

  float gate[8];
#pragma unroll
  for (int j = 0; j < 8; ++j) {
    const int idx = tid + 256 * j;
    gate[j] = ((iv[j] - mi) * ri) * lni_g[idx] + lni_b[idx]
            + ((hv[j] - mh) * rh) * lnh_g[idx] + lnh_b[idx];
  }

  // gate[j] holds idx = tid + 256*j:  i:{0,1}  f:{2,3}  g:{4,5}  o:{6,7}
  float cn[2], oo[2];
#pragma unroll
  for (int jj = 0; jj < 2; ++jj) {
    const int p  = tid + 256 * jj;
    const float i_ = sigmoidf(gate[jj]);
    const float f_ = sigmoidf(gate[2 + jj]);
    const float g_ = tanhf(gate[4 + jj]);
    oo[jj] = sigmoidf(gate[6 + jj]);
    cn[jj] = f_ * cbuf[b * GK + p] + i_ * g_;
  }
  const float mc = block_sum(cn[0] + cn[1]) * (1.0f / GK);
  const float d0 = cn[0] - mc, d1 = cn[1] - mc;
  const float rc = rsqrtf(block_sum(d0 * d0 + d1 * d1) * (1.0f / GK) + LN_EPS);
#pragma unroll
  for (int jj = 0; jj < 2; ++jj) {
    const int p = tid + 256 * jj;
    const float cnew = (cn[jj] - mc) * rc * lnc_g[p] + lnc_b[p];
    cbuf[b * GK + p] = cnew;
    const float h = oo[jj] * tanhf(cnew);
    hbuf[b * GK + p] = h;
    hbf[b * GK + p]  = (__bf16)h;   // bf16 copy feeds the next recurrent GEMM
    if (hx) hx[(long)b * hx_stride + p] = h;
  }
}

// ---------------------------------------------------------------------------
// Host orchestration (graph-capture safe: only async launches/memset/memcpy)
// ---------------------------------------------------------------------------
extern "C" void kernel_launch(void* const* d_in, const int* in_sizes, int n_in,
                              void* d_out, int out_size, void* d_ws, size_t ws_size,
                              hipStream_t stream) {
  const float* x     = (const float*)d_in[0];  // [64,1024,512]
  const float* Wih   = (const float*)d_in[1];  // [2,2048,512]
  const float* Whh   = (const float*)d_in[2];  // [2,2048,512]
  const float* lni_g = (const float*)d_in[3];
  const float* lni_b = (const float*)d_in[4];
  const float* lnh_g = (const float*)d_in[5];
  const float* lnh_b = (const float*)d_in[6];
  const float* lnc_g = (const float*)d_in[7];
  const float* lnc_b = (const float*)d_in[8];
  float* out = (float*)d_out;

  const int B = 64, T = 1024, D = 512, H = 512;
  const int Tc = 64;                 // timestep chunk for bulk input-gate GEMM
  const int WEL = 2 * GN * GK;       // elements per weight tensor (both layers)

  // Workspace layout. fp32 region ~35.7MB, then bf16 region ~8.5MB.
  float* ws = (float*)d_ws;
  size_t off = 0;
  float* ig_chunk = ws + off; off += (size_t)Tc * B * GN;  // [Tc,B,2048]
  float* hg0      = ws + off; off += (size_t)B * GN;       // [64,2048]
  float* ig1      = ws + off; off += (size_t)B * GN;
  float* hg1      = ws + off; off += (size_t)B * GN;
  float* h0       = ws + off; off += (size_t)B * H;        // h0,c0,h1,c1 contiguous
  float* c0       = ws + off; off += (size_t)B * H;
  float* h1       = ws + off; off += (size_t)B * H;
  float* c1       = ws + off; off += (size_t)B * H;
  __bf16* bws = (__bf16*)(ws + off);
  size_t boff = 0;
  __bf16* wih_bf = bws + boff; boff += (size_t)WEL;        // [2,2048,512] bf16
  __bf16* whh_bf = bws + boff; boff += (size_t)WEL;
  __bf16* h0b    = bws + boff; boff += (size_t)B * H;      // h0b,h1b contiguous
  __bf16* h1b    = bws + boff; boff += (size_t)B * H;

  hipMemsetAsync(h0,  0, sizeof(float)  * 4 * B * H, stream);  // h0,c0,h1,c1
  hipMemsetAsync(h0b, 0, sizeof(__bf16) * 2 * B * H, stream);  // h0b,h1b

  const dim3 blk(256);

  // One-time per-call weight conversion to bf16 (both layers each).
  f32_to_bf16_kernel<<<WEL / 8 / 256, blk, 0, stream>>>(Wih, wih_bf, WEL);
  f32_to_bf16_kernel<<<WEL / 8 / 256, blk, 0, stream>>>(Whh, whh_bf, WEL);

  const int stepWaves = (B / 16) * (GN / 64);                 // 128 waves
  const dim3 stepGrid1((stepWaves + 7) / 8, 1);               // 16 blocks
  const dim3 stepGrid2((stepWaves + 7) / 8, 2);               // paired GEMMs

  const __bf16* Wih0b = wih_bf;
  const __bf16* Wih1b = wih_bf + (size_t)GN * D;
  const __bf16* Whh0b = whh_bf;
  const __bf16* Whh1b = whh_bf + (size_t)GN * H;

  for (int t0c = 0; t0c < T; t0c += Tc) {
    // Bulk (parallel) input-gate GEMM for layer 0 over Tc timesteps:
    // ig_chunk[(t-t0c)*B+b, :] = x[b,t,:] @ Wih0^T   (fp32 A, inline cvt)
    const int Mb = Tc * B;
    const int bulkWaves = (Mb / 16) * (GN / 64);
    wmma_gemm_kernel<false><<<dim3((bulkWaves + 7) / 8, 1), blk, 0, stream>>>(
        x, Wih0b, ig_chunk, x, Wih0b, ig_chunk, Mb, B, T * D, D, t0c);

    for (int tt = 0; tt < Tc; ++tt) {
      const int t = t0c + tt;
      // Layer 0 recurrent GEMM: hg0 = h0 @ Whh0^T (pure bf16 operands)
      wmma_gemm_kernel<true><<<stepGrid1, blk, 0, stream>>>(
          h0b, Whh0b, hg0, h0b, Whh0b, hg0, B, B, H, 0, 0);
      // Layer 0 cell (fused LN+gates+cellLN)
      cell_kernel<<<B, blk, 0, stream>>>(
          ig_chunk + (size_t)tt * B * GN, hg0,
          lni_g, lni_b, lnh_g, lnh_b, lnc_g, lnc_b,
          c0, h0, h0b, nullptr, 0);
      // Layer 1 GEMM pair (independent → one launch, gridDim.y=2):
      //   y=0: ig1 = h0 @ Wih1^T     y=1: hg1 = h1 @ Whh1^T
      wmma_gemm_kernel<true><<<stepGrid2, blk, 0, stream>>>(
          h0b, Wih1b, ig1, h1b, Whh1b, hg1, B, B, H, 0, 0);
      // Layer 1 cell; writes h directly into out[b, t, :]
      cell_kernel<<<B, blk, 0, stream>>>(
          ig1, hg1,
          lni_g + GN, lni_b + GN, lnh_g + GN, lnh_b + GN, lnc_g + H, lnc_b + H,
          c1, h1, h1b, out + (size_t)t * H, (long)T * H);
    }
  }

  // Tail of d_out: h stack [L,B,H] then c stack [L,B,H]
  const size_t OUTE = (size_t)B * T * H;
  const size_t BH   = (size_t)B * H;
  hipMemcpyAsync(out + OUTE,          h0, BH * sizeof(float), hipMemcpyDeviceToDevice, stream);
  hipMemcpyAsync(out + OUTE + BH,     h1, BH * sizeof(float), hipMemcpyDeviceToDevice, stream);
  hipMemcpyAsync(out + OUTE + 2 * BH, c0, BH * sizeof(float), hipMemcpyDeviceToDevice, stream);
  hipMemcpyAsync(out + OUTE + 3 * BH, c1, BH * sizeof(float), hipMemcpyDeviceToDevice, stream);
}
